// GATLiteLayer_36180804501652
// MI455X (gfx1250) — compile-verified, matrix operand order checked by reading, not run on hardware
//
#include <hip/hip_runtime.h>
#include <hip/hip_bf16.h>
#include <math.h>

#define N_NODES 8192
#define N_EDGES 262144
#define DIM 256

typedef __attribute__((ext_vector_type(16))) __bf16        v16bf;
typedef __attribute__((ext_vector_type(8)))  float         v8f;
typedef __attribute__((ext_vector_type(4)))  unsigned int  u32x4;
typedef __attribute__((ext_vector_type(8)))  int           i32x8;
typedef __attribute__((ext_vector_type(4)))  int           i32x4;

// ---------------------------------------------------------------------------
// Kernel 1: h = x @ W^T  (x: N x 256 f32, W: 256 x 256 f32 row-major (out,in))
// Workgroup = 4 waves = 64x16 output tile. The block's 16x256 f32 W slab is
// staged into LDS once by the Tensor Data Mover (tensor_load_to_lds +
// s_wait_tensorcnt), then every wave reads B fragments from LDS while the
// streaming x operand comes from global (b128 loads + global_prefetch_b8).
// WMMA fragment layouts follow CDNA5 ISA 7.12.2 (wave32):
//   A (16x32 bf16): lane L holds row m=L%16; element j holds
//       k = 16*(j/8) + 8*(L/16) + (j%8)
//   B (32x16 bf16): lane L holds col n=L%16; same K mapping
//   D (16x16 f32):  VGPR r of lane L holds D[m][n], m = r + 8*(L/16), n=L%16
// ---------------------------------------------------------------------------
__global__ void __launch_bounds__(128)
gat_gemm_h(const float* __restrict__ x, const float* __restrict__ W,
           float* __restrict__ h) {
  __shared__ float lds_w[16 * DIM];   // 16 rows of W = 16 output cols, 16 KB

  const int wave  = threadIdx.x >> 5;
  const int lane  = threadIdx.x & 31;
  const int m0    = (blockIdx.x << 6) + (wave << 4);
  const int n0    = blockIdx.y << 4;
  const int idx16 = lane & 15;
  const int khalf = (lane >> 4) << 3;      // 0 or 8

  // ---- TDM: DMA W[n0..n0+15, 0..255] (f32, 16 KB) into LDS (wave 0 only) ----
  if (threadIdx.x < 32) {
    const unsigned long long gaddr =
        (unsigned long long)(const void*)(W + (size_t)n0 * DIM);
    const unsigned int lds_base =
        (unsigned int)(unsigned long long)(const void*)lds_w; // LDS = addr[31:0]

    union { unsigned long long q[2]; u32x4 v; } g0;
    g0.q[0] = 1ull                                  // count = 1 valid descriptor
            | ((unsigned long long)lds_base << 32); // lds_addr [63:32]
    g0.q[1] = (gaddr & 0x01FFFFFFFFFFFFFFull)       // global_addr [120:64]
            | (2ull << 62);                         // type = 2 ("image") [127:126]

    union { unsigned long long q[4]; i32x8 v; } g1;
    g1.q[0] = (2ull << 16)                          // data_size = 4 bytes
            | ((unsigned long long)DIM << 48);      // tensor_dim0 = 256 (low 16)
    g1.q[1] = (16ull << 16)                         // tensor_dim1 = 16 [111:80]
            | ((unsigned long long)DIM << 48);      // tile_dim0 = 256 [127:112]
    g1.q[2] = 16ull                                 // tile_dim1 = 16 [143:128]
            | ((unsigned long long)DIM << 32);      // tensor_dim0_stride [207:160]
    g1.q[3] = 0;                                    // tensor_dim1_stride = 0

    const i32x4 gz4 = {0, 0, 0, 0};                 // 2D tensor: groups 2/3 unused
    const i32x8 gz8 = {0, 0, 0, 0, 0, 0, 0, 0};
    __builtin_amdgcn_tensor_load_to_lds(g0.v, g1.v, gz4, gz4, gz8, 0);
    __builtin_amdgcn_s_wait_tensorcnt(0);
  }
  __syncthreads();

  // ---- K loop: 8 x wmma_f32_16x16x32_bf16 ----
  const float* xrow = x + (size_t)(m0 + idx16) * DIM + khalf;        // A (global)
  const float* brow = lds_w + (size_t)idx16 * DIM + khalf;           // B (LDS)

  v8f acc = {};
  for (int k0 = 0; k0 < DIM; k0 += 32) {
    __builtin_prefetch(xrow + k0 + 32, 0, 0);       // global_prefetch_b8

    v16bf a, b;
#pragma unroll
    for (int j = 0; j < 16; ++j) {
      const int koff = k0 + ((j >> 3) << 4) + (j & 7);  // 16*(j/8) + j%8
      a[j] = (__bf16)xrow[koff];
      b[j] = (__bf16)brow[koff];                        // ds_load path
    }
    // (neg_a, A, neg_b, B, c_mod, C, reuse_a, reuse_b)
    acc = __builtin_amdgcn_wmma_f32_16x16x32_bf16(
        false, a, false, b, (short)0, acc, false, false);
  }

  // ---- scatter D tile ----
  const int mbase = m0 + ((lane >> 4) << 3);
#pragma unroll
  for (int r = 0; r < 8; ++r) {
    h[(size_t)(mbase + r) * DIM + (n0 + idx16)] = acc[r];
  }
}

// ---------------------------------------------------------------------------
// Kernel 2: per-node attention logits  s_src = h@a_src, s_dst = h@a_dst
// ---------------------------------------------------------------------------
__global__ void __launch_bounds__(256)
gat_node_scores(const float* __restrict__ h, const float* __restrict__ a_src,
                const float* __restrict__ a_dst, float* __restrict__ s_src,
                float* __restrict__ s_dst) {
  const int node = blockIdx.x * (blockDim.x >> 5) + (threadIdx.x >> 5);
  const int lane = threadIdx.x & 31;
  const float* row = h + (size_t)node * DIM;

  float ss = 0.f, sd = 0.f;
#pragma unroll
  for (int t = 0; t < DIM / 32; ++t) {
    const int j = lane + 32 * t;
    const float v = row[j];
    ss = fmaf(v, a_src[j], ss);
    sd = fmaf(v, a_dst[j], sd);
  }
#pragma unroll
  for (int off = 16; off > 0; off >>= 1) {
    ss += __shfl_xor(ss, off, 32);
    sd += __shfl_xor(sd, off, 32);
  }
  if (lane == 0) {
    s_src[node] = ss;
    s_dst[node] = sd;
  }
}

// ---------------------------------------------------------------------------
// Kernel 3: edge logits -> exp, accumulate per-dst softmax denominator.
// ---------------------------------------------------------------------------
__global__ void __launch_bounds__(256)
gat_edge_logits(const long long* __restrict__ edge_index,
                const float* __restrict__ s_src, const float* __restrict__ s_dst,
                float* __restrict__ e_exp, float* __restrict__ denom) {
  const int e = blockIdx.x * blockDim.x + threadIdx.x;
  if (e >= N_EDGES) return;
  const long long src = edge_index[e];
  const long long dst = edge_index[N_EDGES + e];
  float v = s_src[src] + s_dst[dst];
  v = (v > 0.f) ? v : 0.2f * v;            // leaky_relu(0.2)
  const float ex = __expf(v);
  e_exp[e] = ex;
  atomicAdd(&denom[dst], ex);              // global_atomic_add_f32
}

// ---------------------------------------------------------------------------
// Kernel 4: out[dst,:] += (exp(e)/denom[dst]) * h[src,:]   (one wave / edge)
// ---------------------------------------------------------------------------
__global__ void __launch_bounds__(256)
gat_edge_aggregate(const long long* __restrict__ edge_index,
                   const float* __restrict__ e_exp,
                   const float* __restrict__ denom,
                   const float* __restrict__ h, float* __restrict__ out_acc) {
  const int e = blockIdx.x * (blockDim.x >> 5) + (threadIdx.x >> 5);
  if (e >= N_EDGES) return;
  const int lane = threadIdx.x & 31;
  const long long src = edge_index[e];
  const long long dst = edge_index[N_EDGES + e];
  const float w = e_exp[e] / denom[dst];
  if (w <= 1e-6f) return;                  // attn > 1e-6 threshold (wave-uniform)

  const float* hs = h + (size_t)src * DIM;
  float* od = out_acc + (size_t)dst * DIM;
#pragma unroll
  for (int t = 0; t < DIM / 32; ++t) {
    const int j = lane + 32 * t;
    atomicAdd(&od[j], w * hs[j]);
  }
}

// ---------------------------------------------------------------------------
// Kernel 5: z = elu(out) + h; LayerNorm(eps=1e-5) * gamma + beta
// ---------------------------------------------------------------------------
__global__ void __launch_bounds__(256)
gat_epilogue(const float* __restrict__ out_acc, const float* __restrict__ h,
             const float* __restrict__ gamma, const float* __restrict__ beta,
             float* __restrict__ y) {
  const int node = blockIdx.x * (blockDim.x >> 5) + (threadIdx.x >> 5);
  const int lane = threadIdx.x & 31;
  const float* o  = out_acc + (size_t)node * DIM;
  const float* hr = h + (size_t)node * DIM;

  float z[DIM / 32];
  float sum = 0.f, sumsq = 0.f;
#pragma unroll
  for (int t = 0; t < DIM / 32; ++t) {
    const int j = lane + 32 * t;
    const float v = o[j];
    const float elu = (v > 0.f) ? v : (__expf(v) - 1.f);
    const float zz = elu + hr[j];
    z[t] = zz;
    sum += zz;
    sumsq = fmaf(zz, zz, sumsq);
  }
#pragma unroll
  for (int off = 16; off > 0; off >>= 1) {
    sum   += __shfl_xor(sum, off, 32);
    sumsq += __shfl_xor(sumsq, off, 32);
  }
  const float mean = sum * (1.f / DIM);
  const float var  = sumsq * (1.f / DIM) - mean * mean;
  const float rstd = rsqrtf(var + 1e-5f);

  float* yr = y + (size_t)node * DIM;
#pragma unroll
  for (int t = 0; t < DIM / 32; ++t) {
    const int j = lane + 32 * t;
    yr[j] = fmaf(gamma[j], (z[t] - mean) * rstd, beta[j]);
  }
}

// ---------------------------------------------------------------------------
// Launcher. Inputs: x, edge_index, W, a_src, a_dst, gamma, beta.
// Workspace layout: [ h | s_src | s_dst | e_exp | denom | out_acc ]
// ---------------------------------------------------------------------------
extern "C" void kernel_launch(void* const* d_in, const int* in_sizes, int n_in,
                              void* d_out, int out_size, void* d_ws, size_t ws_size,
                              hipStream_t stream) {
  const float*     x          = (const float*)d_in[0];
  const long long* edge_index = (const long long*)d_in[1];   // int64 per reference
  const float*     W          = (const float*)d_in[2];
  const float*     a_src      = (const float*)d_in[3];
  const float*     a_dst      = (const float*)d_in[4];
  const float*     gamma      = (const float*)d_in[5];
  const float*     beta       = (const float*)d_in[6];
  float*           y          = (float*)d_out;

  char* ws = (char*)d_ws;
  float* h       = (float*)ws;                      ws += (size_t)N_NODES * DIM * 4;
  float* s_src   = (float*)ws;                      ws += (size_t)N_NODES * 4;
  float* s_dst   = (float*)ws;                      ws += (size_t)N_NODES * 4;
  float* e_exp   = (float*)ws;                      ws += (size_t)N_EDGES * 4;
  float* denom   = (float*)ws;                      ws += (size_t)N_NODES * 4;
  float* out_acc = (float*)ws;                      ws += (size_t)N_NODES * DIM * 4;

  // zero the atomic accumulators (denom + out_acc are contiguous)
  (void)hipMemsetAsync(denom, 0,
                       ((size_t)N_NODES + (size_t)N_NODES * DIM) * 4, stream);

  // 1) h = x @ W^T via bf16 WMMA (TDM-staged W slab in LDS)
  dim3 ggrid(N_NODES / 64, DIM / 16);
  gat_gemm_h<<<ggrid, 128, 0, stream>>>(x, W, h);

  // 2) node attention scores (8 waves / block)
  gat_node_scores<<<N_NODES / 8, 256, 0, stream>>>(h, a_src, a_dst, s_src, s_dst);

  // 3) edge logits + denominators
  gat_edge_logits<<<N_EDGES / 256, 256, 0, stream>>>(edge_index, s_src, s_dst,
                                                     e_exp, denom);

  // 4) sparse weighted aggregation (one wave per edge)
  gat_edge_aggregate<<<N_EDGES / 8, 256, 0, stream>>>(edge_index, e_exp, denom,
                                                      h, out_acc);

  // 5) ELU + residual + LayerNorm
  gat_epilogue<<<N_NODES / 8, 256, 0, stream>>>(out_acc, h, gamma, beta, y);
}
